// MultiHeadAttention_52484500357538
// MI455X (gfx1250) — matrix operand order repaired and checked by reading, not compiled, not run on hardware
//
#include <hip/hip_runtime.h>
#include <hip/hip_bf16.h>

// ---------------- problem constants ----------------
#define S_LEN   4096
#define DMODEL  768
#define NHEADS  12
#define HDIM    64

// ---------------- vector types ----------------
typedef _Float16 h8   __attribute__((ext_vector_type(8)));
typedef _Float16 v16h __attribute__((ext_vector_type(16)));
typedef float    v8f  __attribute__((ext_vector_type(8)));

// ---------------- WMMA wrapper ----------------
__device__ __forceinline__ v8f wmma_f32(v16h a, v16h b, v8f c) {
    // (neg_a, A, neg_b, B, c_mod, C, reuse_a, reuse_b)
    return __builtin_amdgcn_wmma_f32_16x16x32_f16(false, a, false, b, (short)0, c,
                                                  false, false);
}

// ---------------- 16-lane-group max reduction ----------------
// gfx1250 is gfx12-family: v_permlane16_b32 does an arbitrary permute within
// each 16-lane half -> XOR butterfly with NO ds ops / NO dscnt waits.
#if __has_builtin(__builtin_amdgcn_permlane16)
__device__ __forceinline__ float grp16_max(float v) {
    float t;
    t = __uint_as_float(__builtin_amdgcn_permlane16(
        __float_as_uint(v), __float_as_uint(v), 0x67452301u, 0xEFCDAB89u,
        false, false));                                  // sel = lane ^ 1
    v = fmaxf(v, t);
    t = __uint_as_float(__builtin_amdgcn_permlane16(
        __float_as_uint(v), __float_as_uint(v), 0x54761032u, 0xDCFE98BAu,
        false, false));                                  // sel = lane ^ 2
    v = fmaxf(v, t);
    t = __uint_as_float(__builtin_amdgcn_permlane16(
        __float_as_uint(v), __float_as_uint(v), 0x32107654u, 0xBA98FEDCu,
        false, false));                                  // sel = lane ^ 4
    v = fmaxf(v, t);
    t = __uint_as_float(__builtin_amdgcn_permlane16(
        __float_as_uint(v), __float_as_uint(v), 0xFEDCBA98u, 0x76543210u,
        false, false));                                  // sel = lane ^ 8
    v = fmaxf(v, t);
    return v;
}
#else
__device__ __forceinline__ float grp16_max(float v) {
    v = fmaxf(v, __shfl_xor(v, 1, 32));
    v = fmaxf(v, __shfl_xor(v, 2, 32));
    v = fmaxf(v, __shfl_xor(v, 4, 32));
    v = fmaxf(v, __shfl_xor(v, 8, 32));
    return v;
}
#endif

// ---------------- fragment loaders ----------------
// A fragment: 16x32 f16 tile, row-major source with leading dim `ld`.
// ISA layout: lanes 0-15 -> M=lane, K = k0 + {0..7, 16..23}
//             lanes 16-31 -> M=lane-16, K = k0 + {8..15, 24..31}
// -> per lane two contiguous 16B loads.
__device__ __forceinline__ v16h load_frag_a(const _Float16* base, int ld,
                                            int m0, int k0, int lane) {
    const _Float16* p = base + (long)(m0 + (lane & 15)) * ld
                             + k0 + ((lane >> 4) << 3);
    h8 lo = *(const h8*)(p);
    h8 hi = *(const h8*)(p + 16);
    return __builtin_shufflevector(lo, hi, 0, 1, 2, 3, 4, 5, 6, 7,
                                            8, 9, 10, 11, 12, 13, 14, 15);
}

// B fragment (32x16, K x N) loaded from B^T stored row-major (rows = N,
// contiguous along K) with leading dim `ld`.
// ISA layout: lanes 0-15 -> N=lane, K=k0+{0..7, 16..23} ; lanes 16-31 -> N=lane-16, K=k0+{8..15, 24..31}
// -> per lane two contiguous 16B loads.
__device__ __forceinline__ v16h load_frag_bt(const _Float16* base, int ld,
                                             int n0, int k0, int lane) {
    const _Float16* p = base + (long)(n0 + (lane & 15)) * ld
                             + k0 + ((lane >> 4) << 3);
    h8 lo = *(const h8*)(p);
    h8 hi = *(const h8*)(p + 16);
    return __builtin_shufflevector(lo, hi, 0, 1, 2, 3, 4, 5, 6, 7,
                                            8, 9, 10, 11, 12, 13, 14, 15);
}

// ---------------- f32 -> f16 conversion ----------------
__global__ void cvt_f32_to_f16(const float* __restrict__ src,
                               _Float16* __restrict__ dst, int n) {
    int i = blockIdx.x * blockDim.x + threadIdx.x;
    if (i < n) dst[i] = (_Float16)src[i];
}

// ---------------- WMMA GEMM:  out = (A @ W^T + bias) * scale ----------------
// A: [M][K] f16 row-major.  W: [N][K] f16 row-major (== B^T).  bias: [N] f32.
// MODE 1: f16 out, per-head layout  [H][M][64]   (Q, K)
// MODE 2: f16 out, head-transposed  [H][64][M]   (V^T)
// MODE 3: f32 out, row-major        [M][N]       (final projection)
// Workgroup: 256 threads = 8 waves; WG tile 128x128; wave tile 32x64.
template <int MODE>
__global__ __launch_bounds__(256) void gemm_wmma(
    const _Float16* __restrict__ A, const _Float16* __restrict__ W,
    const float* __restrict__ bias, void* __restrict__ out,
    int M, int N, int K, float scale) {

    const int wave = threadIdx.x >> 5;
    const int lane = threadIdx.x & 31;
    const int wr = wave >> 1;      // 0..3
    const int wc = wave & 1;       // 0..1
    const int m0 = blockIdx.x * 128 + wr * 32;
    const int n0 = blockIdx.y * 128 + wc * 64;

    v8f acc[2][4] = {};

    for (int k0 = 0; k0 < K; k0 += 32) {
        // prefetch next k-slice of A (streams from HBM/L2) -> global_prefetch_b8
        int k1 = (k0 + 32 < K) ? k0 + 32 : k0;
        __builtin_prefetch(A + (long)(m0 + (lane & 15)) * K + k1, 0, 1);
        __builtin_prefetch(A + (long)(m0 + 16 + (lane & 15)) * K + k1, 0, 1);

        v16h a0 = load_frag_a(A, K, m0,      k0, lane);
        v16h a1 = load_frag_a(A, K, m0 + 16, k0, lane);
        v16h b[4];
#pragma unroll
        for (int j = 0; j < 4; ++j)
            b[j] = load_frag_bt(W, K, n0 + 16 * j, k0, lane);
#pragma unroll
        for (int j = 0; j < 4; ++j) {
            acc[0][j] = wmma_f32(a0, b[j], acc[0][j]);
            acc[1][j] = wmma_f32(a1, b[j], acc[1][j]);
        }
    }

    // epilogue: bias + scale + scatter store
    const int rowoff = (lane >> 4) << 3;  // 0 or 8
    const int cl = lane & 15;
#pragma unroll
    for (int i = 0; i < 2; ++i) {
#pragma unroll
        for (int j = 0; j < 4; ++j) {
#pragma unroll
            for (int r = 0; r < 8; ++r) {
                int m = m0 + i * 16 + r + rowoff;
                int n = n0 + j * 16 + cl;
                float v = (acc[i][j][r] + bias[n]) * scale;
                if (MODE == 3) {
                    ((float*)out)[(long)m * N + n] = v;
                } else if (MODE == 1) {
                    int h = n >> 6;
                    ((_Float16*)out)[((long)h * M + m) * HDIM + (n & 63)] =
                        (_Float16)v;
                } else {  // MODE == 2: [H][64][M]
                    int h = n >> 6;
                    ((_Float16*)out)[((long)h * HDIM + (n & 63)) * M + m] =
                        (_Float16)v;
                }
            }
        }
    }
}

// ---------------- flash attention ----------------
// qh/kh: [H][S][64] f16 (q pre-scaled by 1/8). vT: [H][64][S] f16.
// oh: [S][D] f16 (heads merged).
// Grid: (S/64, H), block = 128 (4 waves); each wave owns 16 query rows.
__global__ __launch_bounds__(128) void flash_attn(
    const _Float16* __restrict__ qh, const _Float16* __restrict__ kh,
    const _Float16* __restrict__ vT, _Float16* __restrict__ oh) {

    __shared__ __align__(16) _Float16 pS[4][16 * HDIM];  // wave-private P tiles

    const int wave = threadIdx.x >> 5;
    const int lane = threadIdx.x & 31;
    const int h = blockIdx.y;
    const int qbase = blockIdx.x * 64 + wave * 16;

    const _Float16* Q = qh + (long)h * S_LEN * HDIM;
    const _Float16* K = kh + (long)h * S_LEN * HDIM;
    const _Float16* V = vT + (long)h * HDIM * S_LEN;  // [64][S]

    // Q fragments (persist across key loop): hd slices 0..31, 32..63
    v16h qa0 = load_frag_a(Q, HDIM, qbase, 0, lane);
    v16h qa1 = load_frag_a(Q, HDIM, qbase, 32, lane);

    // all-ones B fragment: row sums of P via WMMA (replaces lane reductions)
    v16h vone;
#pragma unroll
    for (int i = 0; i < 16; ++i) vone[i] = (_Float16)1.0f;

    v8f acc[4] = {};                 // 16 rows x 64 hd output accumulator
    float mrow[8], lrow[8];
#pragma unroll
    for (int r = 0; r < 8; ++r) { mrow[r] = -1e30f; lrow[r] = 0.0f; }

    _Float16* pw = pS[wave];
    const int rowoff = (lane >> 4) << 3;
    const int cl = lane & 15;

    for (int kb = 0; kb < S_LEN; kb += 64) {
        // prefetch next key block's K rows and V^T rows -> global_prefetch_b8
        int kn = (kb + 64 < S_LEN) ? kb + 64 : kb;
        __builtin_prefetch(K + (long)(kn + lane) * HDIM, 0, 1);
        __builtin_prefetch(K + (long)(kn + 32 + lane) * HDIM, 0, 1);
        __builtin_prefetch(V + (long)lane * S_LEN + kn, 0, 1);
        __builtin_prefetch(V + (long)(lane + 32) * S_LEN + kn, 0, 1);

        // ---- scores S = Q K^T (already scaled by 1/8 via Q) ----
        v8f s[4] = {};
#pragma unroll
        for (int j = 0; j < 4; ++j) {
            v16h b0 = load_frag_bt(K, HDIM, kb + 16 * j, 0, lane);
            v16h b1 = load_frag_bt(K, HDIM, kb + 16 * j, 32, lane);
            s[j] = wmma_f32(qa0, b0, s[j]);
            s[j] = wmma_f32(qa1, b1, s[j]);
        }

        // ---- block row-max: pure-VALU permlane16 butterfly per row ----
        float bm[8];
#pragma unroll
        for (int r = 0; r < 8; ++r) {
            float v = fmaxf(fmaxf(s[0][r], s[1][r]), fmaxf(s[2][r], s[3][r]));
            bm[r] = grp16_max(v);
        }

        // ---- online softmax rescale + exponentiate ----
        float alpha[8];
#pragma unroll
        for (int r = 0; r < 8; ++r) {
            float mn = fmaxf(mrow[r], bm[r]);
            alpha[r] = __expf(mrow[r] - mn);
            mrow[r] = mn;
#pragma unroll
            for (int j = 0; j < 4; ++j) {
                float p = __expf(s[j][r] - mn);
                s[j][r] = p;
            }
#pragma unroll
            for (int j = 0; j < 4; ++j) acc[j][r] *= alpha[r];
        }

        // ---- P: C-layout -> A-layout via wave-private LDS tile ----
#pragma unroll
        for (int j = 0; j < 4; ++j)
#pragma unroll
            for (int r = 0; r < 8; ++r)
                pw[(r + rowoff) * HDIM + j * 16 + cl] = (_Float16)s[j][r];
        // wave-local RAW on LDS; DS ops are in-order per wave (compiler waits)

        v16h pa0 = load_frag_a(pw, HDIM, 0, 0, lane);
        v16h pa1 = load_frag_a(pw, HDIM, 0, 32, lane);

        // ---- row sums of P via WMMA with all-ones B (lands replicated in
        //      C-layout: every lane holds its 8 rows' sums) ----
        v8f zero = {};
        v8f sums = wmma_f32(pa0, vone, zero);
        sums = wmma_f32(pa1, vone, sums);
#pragma unroll
        for (int r = 0; r < 8; ++r) lrow[r] = lrow[r] * alpha[r] + sums[r];

        // ---- O += P @ V ; B^T rows are V^T rows (contiguous along keys) ----
#pragma unroll
        for (int j = 0; j < 4; ++j) {
            v16h vb0 = load_frag_bt(V, S_LEN, 16 * j, kb,      lane);
            v16h vb1 = load_frag_bt(V, S_LEN, 16 * j, kb + 32, lane);
            acc[j] = wmma_f32(pa0, vb0, acc[j]);
            acc[j] = wmma_f32(pa1, vb1, acc[j]);
        }
    }

    // ---- normalize and store merged-head output [S][D] f16 ----
#pragma unroll
    for (int r = 0; r < 8; ++r) {
        float inv = 1.0f / lrow[r];
        int m = qbase + r + rowoff;
#pragma unroll
        for (int j = 0; j < 4; ++j) {
            int n = h * HDIM + j * 16 + cl;
            oh[(long)m * DMODEL + n] = (_Float16)(acc[j][r] * inv);
        }
    }
}

// ---------------- workspace layout (bytes) ----------------
#define XH_OFF  0u
#define WQ_OFF  (XH_OFF + S_LEN * DMODEL * 2)           //  6,291,456
#define WK_OFF  (WQ_OFF + DMODEL * DMODEL * 2)
#define WV_OFF  (WK_OFF + DMODEL * DMODEL * 2)
#define WW_OFF  (WV_OFF + DMODEL * DMODEL * 2)
#define QH_OFF  (WW_OFF + DMODEL * DMODEL * 2)
#define KH_OFF  (QH_OFF + S_LEN * DMODEL * 2)
#define VT_OFF  (KH_OFF + S_LEN * DMODEL * 2)
#define OH_OFF  (VT_OFF + S_LEN * DMODEL * 2)
// total = 36,175,872 bytes

extern "C" void kernel_launch(void* const* d_in, const int* in_sizes, int n_in,
                              void* d_out, int out_size, void* d_ws, size_t ws_size,
                              hipStream_t stream) {
    const float* x  = (const float*)d_in[0];
    const float* wq = (const float*)d_in[1];
    const float* bq = (const float*)d_in[2];
    const float* wk = (const float*)d_in[3];
    const float* bk = (const float*)d_in[4];
    const float* wv = (const float*)d_in[5];
    const float* bv = (const float*)d_in[6];
    const float* ww = (const float*)d_in[7];
    const float* bw = (const float*)d_in[8];

    char* ws = (char*)d_ws;
    _Float16* xh  = (_Float16*)(ws + XH_OFF);
    _Float16* wqh = (_Float16*)(ws + WQ_OFF);
    _Float16* wkh = (_Float16*)(ws + WK_OFF);
    _Float16* wvh = (_Float16*)(ws + WV_OFF);
    _Float16* wwh = (_Float16*)(ws + WW_OFF);
    _Float16* qh  = (_Float16*)(ws + QH_OFF);
    _Float16* kh  = (_Float16*)(ws + KH_OFF);
    _Float16* vT  = (_Float16*)(ws + VT_OFF);
    _Float16* oh  = (_Float16*)(ws + OH_OFF);

    const int nx = S_LEN * DMODEL;
    const int nw = DMODEL * DMODEL;
    cvt_f32_to_f16<<<(nx + 255) / 256, 256, 0, stream>>>(x,  xh,  nx);
    cvt_f32_to_f16<<<(nw + 255) / 256, 256, 0, stream>>>(wq, wqh, nw);
    cvt_f32_to_f16<<<(nw + 255) / 256, 256, 0, stream>>>(wk, wkh, nw);
    cvt_f32_to_f16<<<(nw + 255) / 256, 256, 0, stream>>>(wv, wvh, nw);
    cvt_f32_to_f16<<<(nw + 255) / 256, 256, 0, stream>>>(ww, wwh, nw);

    dim3 gproj(S_LEN / 128, DMODEL / 128);  // 32 x 6
    // Q gets 1/sqrt(64) folded into the epilogue
    gemm_wmma<1><<<gproj, 256, 0, stream>>>(xh, wqh, bq, qh, S_LEN, DMODEL,
                                            DMODEL, 0.125f);
    gemm_wmma<1><<<gproj, 256, 0, stream>>>(xh, wkh, bk, kh, S_LEN, DMODEL,
                                            DMODEL, 1.0f);
    gemm_wmma<2><<<gproj, 256, 0, stream>>>(xh, wvh, bv, vT, S_LEN, DMODEL,
                                            DMODEL, 1.0f);

    flash_attn<<<dim3(S_LEN / 64, NHEADS), 128, 0, stream>>>(qh, kh, vT, oh);

    gemm_wmma<3><<<gproj, 256, 0, stream>>>(oh, wwh, bw, d_out, S_LEN, DMODEL,
                                            DMODEL, 1.0f);
}
